// EnergyMACE_2207613190289
// MI455X (gfx1250) — matrix operand order repaired and checked by reading, not compiled, not run on hardware
//
#include <hip/hip_runtime.h>
#include <hip/hip_bf16.h>
#include <math.h>

// ---------------- problem constants ----------------
#define NN   10000
#define EE   160000
#define FF   64
#define SS   5
#define NBES 8
#define RHH  32
#define HIDD 16
#define RMAXF 5.0f
#define AVGF  16.0f

typedef __attribute__((ext_vector_type(16))) __bf16 v16bf;
typedef __attribute__((ext_vector_type(8)))  float  v8f;
typedef int vint4 __attribute__((vector_size(16)));   // native 16B vector for async-LDS builtins

#if defined(__HIP_DEVICE_COMPILE__) && \
    __has_builtin(__builtin_amdgcn_global_load_async_to_lds_b128) && \
    __has_builtin(__builtin_amdgcn_s_wait_asynccnt)
#define ASYNC_LDS 1
#else
#define ASYNC_LDS 0
#endif

__device__ __constant__ int d_LOFLM[16] = {0,1,1,1,2,2,2,2,2,3,3,3,3,3,3,3};

__device__ __forceinline__ __bf16 f2bf(float x) {
  unsigned u = __builtin_bit_cast(unsigned, x);
  unsigned r = (u + 0x7FFFu + ((u >> 16) & 1u)) >> 16;
  unsigned short s = (unsigned short)r;
  return __builtin_bit_cast(__bf16, s);
}

__device__ __forceinline__ float silu(float x) {
  return x * (1.0f / (1.0f + __expf(-x)));
}

__device__ __forceinline__ void async_wait_all() {
#if ASYNC_LDS
  __builtin_amdgcn_s_wait_asynccnt(0);
#endif
}

// ---------------- weight prep: f32 [K][N] -> bf16 transposed [N][K] ----------------
__global__ void k_convT(const float* __restrict__ W, __bf16* __restrict__ BT,
                        int K, int N) {
  int idx = blockIdx.x * 256 + threadIdx.x;
  if (idx >= K * N) return;
  int n = idx / K;
  int k = idx - n * K;
  BT[idx] = f2bf(W[k * N + n]);
}

// ---------------- h0 = one_hot(species) @ W_emb ----------------
__global__ void k_embed(const float* __restrict__ attrs, const float* __restrict__ Wemb,
                        float* __restrict__ h) {
  int idx = blockIdx.x * 256 + threadIdx.x;
  if (idx >= NN * FF) return;
  int n = idx >> 6, f = idx & 63;
  float acc = 0.f;
  for (int s = 0; s < SS; ++s) acc += attrs[n * SS + s] * Wemb[s * FF + f];
  h[idx] = acc;
}

// ---------------- edge geometry: spherical harmonics + bessel radial ----------------
__global__ void k_geom(const float* __restrict__ pos, const float* __restrict__ shifts,
                       const int* __restrict__ snd, const int* __restrict__ rcv,
                       float* __restrict__ sh, float* __restrict__ ef) {
  int e = blockIdx.x * 256 + threadIdx.x;
  if (e >= EE) return;
  int s = snd[e], rv = rcv[e];
  float dx = pos[rv*3+0] - pos[s*3+0] + shifts[e*3+0];
  float dy = pos[rv*3+1] - pos[s*3+1] + shifts[e*3+1];
  float dz = pos[rv*3+2] - pos[s*3+2] + shifts[e*3+2];
  float r = sqrtf(dx*dx + dy*dy + dz*dz + 1e-12f);
  float inv = 1.0f / r;
  float x = dx*inv, y = dy*inv, z = dz*inv;
  const float c3=1.7320508f, c15=3.8729833f, c5=2.2360680f, c105=10.246951f;
  const float c358=2.0916500f, c218=1.6201852f, c7=2.6457513f;
  float* S = sh + (long)e * 16;
  S[0]=1.0f; S[1]=c3*x; S[2]=c3*y; S[3]=c3*z;
  S[4]=c15*x*y; S[5]=c15*y*z; S[6]=0.5f*c5*(3.f*z*z-1.f); S[7]=c15*x*z;
  S[8]=0.5f*c15*(x*x-y*y);
  S[9]=c358*y*(3.f*x*x-y*y); S[10]=c105*x*y*z; S[11]=c218*y*(5.f*z*z-1.f);
  S[12]=0.5f*c7*(5.f*z*z*z-3.f*z); S[13]=c218*x*(5.f*z*z-1.f);
  S[14]=0.5f*c105*z*(x*x-y*y); S[15]=c358*x*(x*x-3.f*y*y);
  float xr = r * (1.0f / RMAXF);
  float x2 = xr*xr, x4 = x2*x2, x5 = x4*xr, x6 = x5*xr, x7 = x6*xr;
  float env = 1.0f - 21.0f*x5 + 35.0f*x6 - 15.0f*x7;
  env = (xr < 1.0f) ? env : 0.0f;
  float pref = 0.63245553f /* sqrt(2/5) */ / fmaxf(r, 1e-6f) * env;
  float* F = ef + (long)e * NBES;
  for (int n = 1; n <= NBES; ++n)
    F[n-1] = pref * __sinf(3.14159265358979f * (float)n * xr);
}

// ---------------- zero fill ----------------
__global__ void k_zero(float* __restrict__ p, long n) {
  long i = (long)blockIdx.x * 256 + threadIdx.x;
  long stride = (long)gridDim.x * 256;
  for (; i < n; i += stride) p[i] = 0.0f;
}

// ============= core: 64-column WMMA GEMM block (128 threads, 64 rows) =============
// C[r,:64] = scale*A[r,:64] @ B  (B given bf16-transposed as BT[64][64]), + Cin.
// A tile is converted to bf16 during staging so fragments are pure ds_load_b128.
__device__ __forceinline__ void gemm64_block(
    const float* __restrict__ A, int strideA,
    const __bf16* __restrict__ BT,
    const float* __restrict__ Cin, int strideCin,
    float* __restrict__ C, int strideC,
    int nrows, float scale) {
  __shared__ alignas(16) __bf16 lA[64 * 64];
  __shared__ alignas(16) __bf16 lB[64 * 64];
  int tid = threadIdx.x;                    // 128 threads = 4 waves
  int rowBase = blockIdx.x * 64;
  bool full = (rowBase + 64) <= nrows;
#if ASYNC_LDS
  for (int i = tid * 16; i < 64 * 64 * 2; i += 128 * 16)
    __builtin_amdgcn_global_load_async_to_lds_b128(
        (vint4*)((char*)BT + i), (vint4*)((char*)lB + i), 0, 0);
#else
  for (int i = tid; i < 512; i += 128)
    ((uint4*)lB)[i] = ((const uint4*)BT)[i];
#endif
  if (full) {
    for (int i = tid; i < 64 * 64; i += 128) {
      int rr = i >> 6, cc = i & 63;
      lA[i] = f2bf(A[(long)(rowBase + rr) * strideA + cc] * scale);
    }
  } else {
    for (int i = tid; i < 64 * 64; i += 128) {
      int rr = i >> 6, cc = i & 63;
      int gr = rowBase + rr;
      lA[i] = f2bf(((gr < nrows) ? A[(long)gr * strideA + cc] : 0.0f) * scale);
    }
  }
  async_wait_all();
  __syncthreads();
  int wave = tid >> 5, lane = tid & 31;
  int mlo = lane & 15, mhi = (lane >> 4) & 1;
  const __bf16* wA = &lA[(wave * 16) * 64];
  v8f acc[4];
  if (Cin != nullptr && full) {
#pragma unroll
    for (int nt = 0; nt < 4; ++nt)
      for (int r = 0; r < 8; ++r)
        acc[nt][r] = Cin[(long)(rowBase + wave * 16 + r + mhi * 8) * strideCin +
                         nt * 16 + mlo];
  } else {
#pragma unroll
    for (int nt = 0; nt < 4; ++nt)
      for (int r = 0; r < 8; ++r) {
        int grow = rowBase + wave * 16 + r + mhi * 8;
        float cv = 0.0f;
        if (Cin != nullptr && grow < nrows) cv = Cin[(long)grow * strideCin + nt * 16 + mlo];
        acc[nt][r] = cv;
      }
  }
#pragma unroll
  for (int s = 0; s < 2; ++s) {
    v16bf a;
#pragma unroll
    for (int e = 0; e < 16; ++e) {
      int kk = s * 32 + mhi * 16 + e;                          // A 16x32 16-bit layout (contiguous k)
      a[e] = wA[mlo * 64 + kk];
    }
#pragma unroll
    for (int nt = 0; nt < 4; ++nt) {
      v16bf b;
      int kb = s * 32 + mhi * 16;                              // B 32x16 16-bit layout
#pragma unroll
      for (int e = 0; e < 16; ++e) b[e] = lB[(nt * 16 + mlo) * 64 + kb + e];
      acc[nt] = __builtin_amdgcn_wmma_f32_16x16x32_bf16(
          false, a, false, b, (short)0, acc[nt], false, false);
    }
  }
  if (full) {
#pragma unroll
    for (int nt = 0; nt < 4; ++nt)
      for (int r = 0; r < 8; ++r)
        C[(long)(rowBase + wave * 16 + r + mhi * 8) * strideC + nt * 16 + mlo] =
            acc[nt][r];
  } else {
#pragma unroll
    for (int nt = 0; nt < 4; ++nt)
      for (int r = 0; r < 8; ++r) {
        int grow = rowBase + wave * 16 + r + mhi * 8;
        if (grow < nrows) C[(long)grow * strideC + nt * 16 + mlo] = acc[nt][r];
      }
  }
}

__global__ void __launch_bounds__(128)
k_gemm64(const float* A, int strideA, const __bf16* BT,
         const float* Cin, int strideCin,
         float* C, int strideC, int nrows, float scale) {
  gemm64_block(A, strideA, BT, Cin, strideCin, C, strideC, nrows, scale);
}

// in-place channel mix: A[:,k,:] = (A[:,k,:]/AVG) @ Wmix[L_OF_LM[k]]
__global__ void __launch_bounds__(128)
k_mix(float* Aacc, const __bf16* WmixT, int nrows, float scale) {
  int k = blockIdx.y;
  int l = d_LOFLM[k];
  gemm64_block(Aacc + k * 64, 1024, WmixT + l * 64 * 64,
               nullptr, 0, Aacc + k * 64, 1024, nrows, scale);
}

// ============= fused edge kernel: radial MLP + WMMA [E x 32]@[32 x 256] + scatter ====
__global__ void __launch_bounds__(128)
k_edge(const float* __restrict__ ef, const float* __restrict__ sh,
       const float* __restrict__ hu,
       const int* __restrict__ snd, const int* __restrict__ rcv,
       const float* __restrict__ Wr0, const float* __restrict__ Wr1,
       const __bf16* __restrict__ Wr2T, float* __restrict__ Aacc) {
  __shared__ alignas(16) float  lWr0[8 * 32];
  __shared__ alignas(16) float  lWr1[32 * 32];
  __shared__ alignas(16) __bf16 lWr2T[256 * 32];
  __shared__ alignas(16) float  lHu[64 * 64];
  __shared__ alignas(16) float  lSh[64 * 16];
  __shared__ alignas(16) int    lSnd[64];
  __shared__ alignas(16) int    lRcv[64];
  __shared__ alignas(16) __bf16 lRr[64 * 32];
  int tid = threadIdx.x;                    // 128 threads, 64 edges/block
  int ebase = blockIdx.x * 64;
  // ---- phase 0: ids + small weights (sync copies); big tiles via async DMA ----
  for (int i = tid; i < 8  * 32; i += 128) lWr0[i] = Wr0[i];
  for (int i = tid; i < 32 * 32; i += 128) lWr1[i] = Wr1[i];
  if (tid < 64) {
    int s0 = snd[ebase + tid];
    lSnd[tid] = s0;
    lRcv[tid] = rcv[ebase + tid];
    __builtin_prefetch(hu + (long)s0 * FF, 0, 0);   // global_prefetch_b8
  }
#if ASYNC_LDS
  for (int i = tid * 16; i < 256 * 32 * 2; i += 128 * 16)    // Wr2T: 16 KB
    __builtin_amdgcn_global_load_async_to_lds_b128(
        (vint4*)((char*)Wr2T + i), (vint4*)((char*)lWr2T + i), 0, 0);
  for (int i = tid * 16; i < 64 * 16 * 4; i += 128 * 16)     // sh tile: 4 KB
    __builtin_amdgcn_global_load_async_to_lds_b128(
        (vint4*)((char*)(sh + (long)ebase * 16) + i),
        (vint4*)((char*)lSh + i), 0, 0);
#else
  for (int i = tid; i < 1024; i += 128) ((uint4*)lWr2T)[i] = ((const uint4*)Wr2T)[i];
  for (int i = tid; i < 256;  i += 128) ((uint4*)lSh)[i] = ((const uint4*)(sh + (long)ebase * 16))[i];
#endif
  __syncthreads();                          // lSnd visible for the hu gather
  // ---- phase 1: async-gather hu rows while 64 lanes run the radial MLP ----
#if ASYNC_LDS
  for (int i = tid; i < 64 * 16; i += 128) {                 // 64 rows x 16 chunks(16B)
    int le = i >> 4, ch = i & 15;
    __builtin_amdgcn_global_load_async_to_lds_b128(
        (vint4*)(hu + (long)lSnd[le] * FF + ch * 4),
        (vint4*)(lHu + le * 64 + ch * 4), 0, 0);
  }
#else
  for (int i = tid; i < 64 * 64; i += 128) {
    int le = i >> 6, f = i & 63;
    lHu[i] = hu[(long)lSnd[le] * FF + f];
  }
#endif
  if (tid < 64) {                           // radial MLP 8 -> 32 -> 32 (VALU)
    long e = ebase + tid;
    float efv[NBES];
    for (int b = 0; b < NBES; ++b) efv[b] = ef[e * NBES + b];
    float s1[RHH];
    for (int j = 0; j < RHH; ++j) {
      float acc = 0.f;
      for (int b = 0; b < NBES; ++b) acc += efv[b] * lWr0[b * RHH + j];
      s1[j] = silu(acc);
    }
    for (int j = 0; j < RHH; ++j) {
      float acc = 0.f;
      for (int i2 = 0; i2 < RHH; ++i2) acc += s1[i2] * lWr1[i2 * RHH + j];
      lRr[tid * 32 + j] = f2bf(silu(acc));
    }
  }
  async_wait_all();
  __syncthreads();
  // ---- phase 2: WMMA [16 edges x 32] @ [32 x 256] and scatter with f32 atomics ----
  int wave = tid >> 5, lane = tid & 31;
  int mlo = lane & 15, mhi = (lane >> 4) & 1;
  v16bf a;
#pragma unroll
  for (int e = 0; e < 16; ++e) {
    int kk = mhi * 16 + e;
    a[e] = lRr[(wave * 16 + mlo) * 32 + kk];
  }
  for (int nt = 0; nt < 16; ++nt) {
    v16bf b;
    int kb = mhi * 16;
#pragma unroll
    for (int e2 = 0; e2 < 16; ++e2) b[e2] = lWr2T[(nt * 16 + mlo) * 32 + kb + e2];
    v8f acc = {0.f,0.f,0.f,0.f,0.f,0.f,0.f,0.f};
    acc = __builtin_amdgcn_wmma_f32_16x16x32_bf16(
        false, a, false, b, (short)0, acc, false, false);
    int c = nt * 16 + mlo;                  // column 0..255 of Rfull
    int l = c >> 6, f = c & 63;
    int kbeg = (l == 0) ? 0 : (l == 1) ? 1 : (l == 2) ? 4 : 9;
    int kcnt = 2 * l + 1;
#pragma unroll
    for (int r = 0; r < 8; ++r) {
      int le = wave * 16 + r + mhi * 8;     // local edge of this value
      float v = acc[r] * lHu[le * 64 + f];
      float* dst = Aacc + (long)lRcv[le] * 1024 + f;
      for (int k = 0; k < kcnt; ++k)
        unsafeAtomicAdd(dst + (kbeg + k) * 64, v * lSh[le * 16 + kbeg + k]);
    }
  }
}

// ---------------- combine: B1, B2, B1*B2, and species-selected self-connection ------
__global__ void k_combine(const float* __restrict__ Amix, const float* __restrict__ h,
                          const float* __restrict__ attrs, const float* __restrict__ Wsc,
                          float* __restrict__ B1, float* __restrict__ B2,
                          float* __restrict__ B12, float* __restrict__ hnew) {
  int idx = blockIdx.x * 256 + threadIdx.x;
  if (idx >= NN * FF) return;
  int n = idx >> 6, f = idx & 63;
  const float* An = Amix + (long)n * 1024 + f;
  float b1 = An[0];
  float b2 = 0.f;
  for (int k = 0; k < 16; ++k) { float v = An[k * 64]; b2 += v * v; }
  B1[idx] = b1; B2[idx] = b2; B12[idx] = b2 * b1;
  int sp = 0;
  for (int s = 1; s < SS; ++s) if (attrs[n * SS + s] > 0.5f) sp = s;
  const float* W = Wsc + (long)sp * FF * FF;
  float sc = 0.f;
  for (int g = 0; g < FF; ++g) sc += h[n * FF + g] * W[g * FF + f];
  hnew[idx] = sc;
}

// ---------------- readouts ----------------
__global__ void k_ro0(const float* __restrict__ h, const float* __restrict__ w,
                      float* __restrict__ out) {
  int n = blockIdx.x * 256 + threadIdx.x;
  if (n >= NN) return;
  float acc = 0.f;
  for (int f = 0; f < FF; ++f) acc += h[n * FF + f] * w[f];
  out[2 * n + 0] = acc;
}

__global__ void k_ro1(const float* __restrict__ h, const float* __restrict__ Wa,
                      const float* __restrict__ wb, float* __restrict__ out) {
  int n = blockIdx.x * 256 + threadIdx.x;
  if (n >= NN) return;
  float e1 = 0.f;
  for (int j = 0; j < HIDD; ++j) {
    float t = 0.f;
    for (int f = 0; f < FF; ++f) t += h[n * FF + f] * Wa[f * HIDD + j];
    e1 += silu(t) * wb[j];
  }
  out[2 * n + 1] = e1;
}

// =====================================================================================
extern "C" void kernel_launch(void* const* d_in, const int* in_sizes, int n_in,
                              void* d_out, int out_size, void* d_ws, size_t ws_size,
                              hipStream_t stream) {
  (void)in_sizes; (void)n_in; (void)out_size; (void)ws_size;
  const float* positions = (const float*)d_in[0];
  const float* attrs     = (const float*)d_in[1];
  const float* shifts    = (const float*)d_in[2];
  const int*   senders   = (const int*)d_in[3];
  const int*   receivers = (const int*)d_in[4];
  const float* W_emb     = (const float*)d_in[5];
  const float* Wup[2]  = {(const float*)d_in[6],  (const float*)d_in[15]};
  const float* Wr0[2]  = {(const float*)d_in[7],  (const float*)d_in[16]};
  const float* Wr1[2]  = {(const float*)d_in[8],  (const float*)d_in[17]};
  const float* Wr2[2]  = {(const float*)d_in[9],  (const float*)d_in[18]};
  const float* Wmix[2] = {(const float*)d_in[10], (const float*)d_in[19]};
  const float* Wsc[2]  = {(const float*)d_in[11], (const float*)d_in[20]};
  const float* Wp[2][3] = {
    {(const float*)d_in[12], (const float*)d_in[13], (const float*)d_in[14]},
    {(const float*)d_in[21], (const float*)d_in[22], (const float*)d_in[23]}};
  const float* w_ro0  = (const float*)d_in[24];
  const float* W_ro1a = (const float*)d_in[25];
  const float* w_ro1b = (const float*)d_in[26];
  float* out = (float*)d_out;

  size_t off = 0;
  auto wsalloc = [&](size_t bytes) -> void* {
    void* p = (char*)d_ws + off;
    off += (bytes + 255) & ~(size_t)255;
    return p;
  };
  float* sh   = (float*)wsalloc((size_t)EE * 16 * 4);
  float* efb  = (float*)wsalloc((size_t)EE * NBES * 4);
  float* hA   = (float*)wsalloc((size_t)NN * FF * 4);
  float* hB   = (float*)wsalloc((size_t)NN * FF * 4);
  float* hu   = (float*)wsalloc((size_t)NN * FF * 4);
  float* Aacc = (float*)wsalloc((size_t)NN * 1024 * 4);
  float* B1   = (float*)wsalloc((size_t)NN * FF * 4);
  float* B2   = (float*)wsalloc((size_t)NN * FF * 4);
  float* B12  = (float*)wsalloc((size_t)NN * FF * 4);
  __bf16* WupT[2];  __bf16* WmixT[2]; __bf16* WpT[2][3]; __bf16* Wr2T[2];
  for (int l = 0; l < 2; ++l) {
    WupT[l]  = (__bf16*)wsalloc(64 * 64 * 2);
    WmixT[l] = (__bf16*)wsalloc(4 * 64 * 64 * 2);
    for (int p = 0; p < 3; ++p) WpT[l][p] = (__bf16*)wsalloc(64 * 64 * 2);
    Wr2T[l]  = (__bf16*)wsalloc(256 * 32 * 2);
  }

  auto conv = [&](const float* W, __bf16* BT, int K, int N) {
    int total = K * N;
    k_convT<<<(total + 255) / 256, 256, 0, stream>>>(W, BT, K, N);
  };
  for (int l = 0; l < 2; ++l) {
    conv(Wup[l], WupT[l], 64, 64);
    for (int j = 0; j < 4; ++j) conv(Wmix[l] + j * 4096, WmixT[l] + j * 4096, 64, 64);
    for (int p = 0; p < 3; ++p) conv(Wp[l][p], WpT[l][p], 64, 64);
    conv(Wr2[l], Wr2T[l], 32, 256);
  }

  k_embed<<<(NN * FF + 255) / 256, 256, 0, stream>>>(attrs, W_emb, hA);
  k_geom<<<(EE + 255) / 256, 256, 0, stream>>>(positions, shifts, senders, receivers, sh, efb);

  const int rowBlocks = (NN + 63) / 64;  // 157
  for (int l = 0; l < 2; ++l) {
    float* hin  = (l == 0) ? hA : hB;
    float* hout = (l == 0) ? hB : hA;
    k_zero<<<2048, 256, 0, stream>>>(Aacc, (long)NN * 1024);
    k_gemm64<<<rowBlocks, 128, 0, stream>>>(hin, FF, WupT[l], nullptr, 0, hu, FF, NN, 1.0f);
    k_edge<<<EE / 64, 128, 0, stream>>>(efb, sh, hu, senders, receivers,
                                        Wr0[l], Wr1[l], Wr2T[l], Aacc);
    k_mix<<<dim3(rowBlocks, 16), 128, 0, stream>>>(Aacc, WmixT[l], NN, 1.0f / AVGF);
    k_combine<<<(NN * FF + 255) / 256, 256, 0, stream>>>(Aacc, hin, attrs, Wsc[l],
                                                         B1, B2, B12, hout);
    k_gemm64<<<rowBlocks, 128, 0, stream>>>(B1,  FF, WpT[l][0], hout, FF, hout, FF, NN, 1.0f);
    k_gemm64<<<rowBlocks, 128, 0, stream>>>(B2,  FF, WpT[l][1], hout, FF, hout, FF, NN, 1.0f);
    k_gemm64<<<rowBlocks, 128, 0, stream>>>(B12, FF, WpT[l][2], hout, FF, hout, FF, NN, 1.0f);
    if (l == 0) k_ro0<<<(NN + 255) / 256, 256, 0, stream>>>(hout, w_ro0, out);
    else        k_ro1<<<(NN + 255) / 256, 256, 0, stream>>>(hout, W_ro1a, w_ro1b, out);
  }
}
